// FCNN_40553081209054
// MI455X (gfx1250) — matrix operand order repaired and checked by reading, not compiled, not run on hardware
//
#include <hip/hip_runtime.h>
#include <cstdint>

#define HIDDEN          256
#define N_MID           28
#define ROWS_PER_BLOCK  64    // 4 row-tiles of 16, each computed by a wave PAIR
#define HSTRIDE         264   // halfs per H row (256 + 8 pad) -> conflict-free ds loads
#define WSTRIDE         264   // halfs per W row in LDS (TDM pads 512B rows by 16B)

typedef __attribute__((ext_vector_type(16))) _Float16 v16h;
typedef __attribute__((ext_vector_type(8)))  _Float16 v8h;
typedef __attribute__((ext_vector_type(8)))  float    v8f;
typedef __attribute__((ext_vector_type(4)))  unsigned u32x4;
typedef __attribute__((ext_vector_type(4)))  int      i32x4;
typedef __attribute__((ext_vector_type(8)))  int      i32x8;

__device__ __forceinline__ float fast_sigmoid(float v) {
    return __builtin_amdgcn_rcpf(1.0f + __expf(-v));
}

// Issue one TDM DMA: 256x256 f16 tile, global -> LDS, with 16B LDS pad per
// 512B row so the LDS image has a 264-half row stride.
__device__ __forceinline__ void tdm_load_weights(const _Float16* gsrc, unsigned lds_addr) {
    unsigned long long ga = (unsigned long long)(uintptr_t)gsrc;
    u32x4 g0;
    g0.x = 1u;                                   // count=1 (valid user descriptor)
    g0.y = lds_addr;                             // LDS destination byte address
    g0.z = (unsigned)(ga & 0xFFFFFFFFu);         // global_addr[31:0]
    g0.w = (unsigned)((ga >> 32) & 0x01FFFFFFu)  // global_addr[56:32]
         | (2u << 30);                           // type = 2 ("image")
    i32x8 g1;
    g1[0] = (1 << 16)                            // data_size = 2 bytes
          | (1 << 20)                            // pad_enable
          | (6 << 22)                            // pad_interval = 128 dwords (512B row)
          | (3 << 25);                           // pad_amount  = 4 dwords (16B)
    g1[1] = (int)(256u << 16);                   // tensor_dim0 = 256
    g1[2] = (int)(256u << 16);                   // tensor_dim1 = 256
    g1[3] = (int)(256u << 16);                   // tile_dim0   = 256
    g1[4] = 256;                                 // tile_dim1 = 256, tile_dim2 = 0
    g1[5] = 256;                                 // tensor_dim0_stride = 256 elems
    g1[6] = 0;
    g1[7] = 0;
    i32x4 gz = {0, 0, 0, 0};
#if defined(__clang_major__) && (__clang_major__ >= 23)
    i32x8 gz8 = {0, 0, 0, 0, 0, 0, 0, 0};
    __builtin_amdgcn_tensor_load_to_lds(g0, g1, gz, gz, gz8, 0);
#else
    __builtin_amdgcn_tensor_load_to_lds(g0, g1, gz, gz, 0);
#endif
}

// Wt[l][k][j] = (f16) Wm[l][j][k]   (pre-transposed so B fragments are contiguous)
__global__ __launch_bounds__(256) void fcnn_convert_weights(
    const float* __restrict__ Wm, _Float16* __restrict__ Wt) {
    int idx = blockIdx.x * 256 + threadIdx.x;    // 28*65536 total
    int l   = idx >> 16;
    int rem = idx & 0xFFFF;
    int k   = rem >> 8;
    int j   = rem & 255;
    Wt[idx] = (_Float16)Wm[(l << 16) + (j << 8) + k];
}

__global__ __launch_bounds__(256) void fcnn_fused(
    const float*    __restrict__ x,  const float* __restrict__ y,
    const float*    __restrict__ W1, const float* __restrict__ b1,
    const float*    __restrict__ bm, const float* __restrict__ Wl,
    const float*    __restrict__ bl, const _Float16* __restrict__ Wt,
    float* __restrict__ out) {

    __shared__ _Float16 Hs[ROWS_PER_BLOCK * HSTRIDE];   //  33.8 KB activations
    __shared__ _Float16 Ws[2][HIDDEN * WSTRIDE];        // 270.3 KB double-buffered W

    const int tid  = threadIdx.x;
    const int lane = tid & 31;
    const int wave = tid >> 5;
    const int row0 = blockIdx.x * ROWS_PER_BLOCK;

    // Prefetch layer-0 weights into buffer 0; overlaps with the input layer.
    if (tid < 32) tdm_load_weights(Wt, (unsigned)(uintptr_t)(void*)Ws[0]);

    // ---- Input layer: h = sigmoid(x*W1[:,0] + y*W1[:,1] + b1), K=2 (VALU) ----
    {
        const float w0 = W1[tid * 2 + 0];
        const float w1 = W1[tid * 2 + 1];
        const float b  = b1[tid];
        for (int r = 0; r < ROWS_PER_BLOCK; ++r) {
            float v = x[row0 + r] * w0 + y[row0 + r] * w1 + b;
            Hs[r * HSTRIDE + tid] = (_Float16)fast_sigmoid(v);
        }
    }

    const int mrow  = lane & 15;          // row within the 16-row tile
    const int khalf = (lane >> 4) * 8;    // A-fragment half-lane K offset
    const int wr0   = (wave >> 1) * 16;   // wave PAIR shares one 16-row tile
    const int jbase = (wave & 1) * 8;     // this wave's 8 N-tiles (cols 0-127 / 128-255)

    // ---- 28 middle layers: 256x256 GEMM + sigmoid, fully WMMA ----
    for (int l = 0; l < N_MID; ++l) {
        const _Float16* Wbuf = Ws[l & 1];

        // Wave 0: TDM for layer l is the only outstanding tensor op -> wait it.
        if (tid < 32) __builtin_amdgcn_s_wait_tensorcnt(0);
        __syncthreads();   // weights[l] visible; prev layer's H writes visible;
                           // all readers of buffer (l+1)&1 are provably done.

        // Prefetch layer l+1 into the other buffer; overlaps all compute below.
        if (l + 1 < N_MID && tid < 32)
            tdm_load_weights(Wt + (size_t)(l + 1) * HIDDEN * HIDDEN,
                             (unsigned)(uintptr_t)(void*)Ws[(l + 1) & 1]);

        // Hoist A fragments (this tile's 16 rows of H) into registers.
        v16h A[8];
        {
            const _Float16* hrow = &Hs[(wr0 + mrow) * HSTRIDE];
#pragma unroll
            for (int kt = 0; kt < 8; ++kt) {
                v8h lo = *(const v8h*)(hrow + kt * 32 + khalf);
                v8h hi = *(const v8h*)(hrow + kt * 32 + khalf + 16);
                A[kt]  = __builtin_shufflevector(lo, hi,
                         0, 1, 2, 3, 4, 5, 6, 7, 8, 9, 10, 11, 12, 13, 14, 15);
            }
        }

        // 8 N-tiles of 16 cols per wave, each accumulated over 8 K-steps of 32.
        for (int jt = 0; jt < 8; ++jt) {
            const int j0 = (jbase + jt) * 16;
            const float bias = bm[l * HIDDEN + j0 + mrow];
            v8f acc = {bias, bias, bias, bias, bias, bias, bias, bias};
#pragma unroll
            for (int kt = 0; kt < 8; ++kt) {
                v16h B;
#pragma unroll
                for (int s = 0; s < 16; ++s)
                    B[s] = Wbuf[(kt * 32 + khalf + (s & 7) + ((s >> 3) << 4)) * WSTRIDE + j0 + mrow];
                acc = __builtin_amdgcn_wmma_f32_16x16x32_f16(
                          false, A[kt], false, B, (short)0, acc, false, false);
            }
            // sigmoid + write new H (synced by the next layer's barrier)
            const int col   = j0 + mrow;
            const int rbase = wr0 + ((lane >> 4) << 3);
#pragma unroll
            for (int m = 0; m < 8; ++m) {
                Hs[(rbase + m) * HSTRIDE + col] = (_Float16)fast_sigmoid(acc[m]);
            }
        }
    }

    // ---- Output layer: out = h @ Wl.T + bl, N=1 (VALU dot product) ----
    __syncthreads();
    if (tid < ROWS_PER_BLOCK) {
        float acc = bl[0];
        const _Float16* hrow = &Hs[tid * HSTRIDE];
#pragma unroll 8
        for (int j = 0; j < HIDDEN; ++j) acc += (float)hrow[j] * Wl[j];
        out[row0 + tid] = acc;
    }
}

extern "C" void kernel_launch(void* const* d_in, const int* in_sizes, int n_in,
                              void* d_out, int out_size, void* d_ws, size_t ws_size,
                              hipStream_t stream) {
    const float* x  = (const float*)d_in[0];
    const float* y  = (const float*)d_in[1];
    const float* W1 = (const float*)d_in[2];
    const float* b1 = (const float*)d_in[3];
    const float* Wm = (const float*)d_in[4];
    const float* bm = (const float*)d_in[5];
    const float* Wl = (const float*)d_in[6];
    const float* bl = (const float*)d_in[7];
    float*     out  = (float*)d_out;
    _Float16*  Wt   = (_Float16*)d_ws;       // 28*256*256 halfs = 3.5 MB

    const int n = in_sizes[0];               // 65536 rows

    fcnn_convert_weights<<<(N_MID * HIDDEN * HIDDEN) / 256, 256, 0, stream>>>(Wm, Wt);
    fcnn_fused<<<n / ROWS_PER_BLOCK, 256, 0, stream>>>(
        x, y, W1, b1, bm, Wl, bl, Wt, out);
}
